// LlamaAttention_55027120997085
// MI455X (gfx1250) — matrix operand order repaired and checked by reading, not compiled, not run on hardware
//
#include <hip/hip_runtime.h>
#include <hip/hip_bf16.h>

// ---------- types ----------
typedef __attribute__((ext_vector_type(8)))  float  v8f;
typedef __attribute__((ext_vector_type(16))) __bf16 v16bf;

struct Frag { uint4 u[2]; };                       // 32 bytes = 16 bf16
__device__ __forceinline__ v16bf frag_cast(const Frag& f) {
    return __builtin_bit_cast(v16bf, f);
}

__device__ __forceinline__ unsigned short f2bf(float x) {
    unsigned u = __builtin_bit_cast(unsigned, x);
    u += 0x7FFFu + ((u >> 16) & 1u);               // round-to-nearest-even
    return (unsigned short)(u >> 16);
}

// ---------- CDNA5 async global->LDS copy (ASYNCcnt path) ----------
__device__ __forceinline__ unsigned lds_off(const void* p) {
    return (unsigned)(unsigned long long)p;        // low 32 bits = LDS byte offset
}
__device__ __forceinline__ void async_b128(unsigned ldsaddr, const void* g) {
    unsigned long long a = (unsigned long long)g;
    asm volatile("global_load_async_to_lds_b128 %0, %1, off"
                 :: "v"(ldsaddr), "v"(a) : "memory");
}
__device__ __forceinline__ void wait_async0() {
#if __has_builtin(__builtin_amdgcn_s_wait_asynccnt)
    __builtin_amdgcn_s_wait_asynccnt(0);
#else
    asm volatile("s_wait_asynccnt 0x0" ::: "memory");
#endif
}

// ---------- fp32 -> bf16 elementwise ----------
__global__ __launch_bounds__(256) void cvt_bf16(const float* __restrict__ X,
                                                unsigned short* __restrict__ Y, int n) {
    int i = (blockIdx.x * 256 + threadIdx.x) * 4;
    if (i + 3 < n) {
        float4 v = *(const float4*)(X + i);
        Y[i + 0] = f2bf(v.x); Y[i + 1] = f2bf(v.y);
        Y[i + 2] = f2bf(v.z); Y[i + 3] = f2bf(v.w);
    }
}

// ---------- fused RoPE + fp32 -> bf16 (head_dim 128, rotate-half) ----------
__global__ __launch_bounds__(256) void rope_convert(const float* __restrict__ X,
                                                    unsigned short* __restrict__ Y,
                                                    int S, int HD) {
    int idx = blockIdx.x * 256 + threadIdx.x;
    int halfcols = HD >> 1;
    if (idx >= S * halfcols) return;
    int s = idx / halfcols;
    int c = idx - s * halfcols;
    int head = c >> 6;
    int d    = c & 63;
    float inv = __powf(10000.0f, -(float)(2 * d) * (1.0f / 128.0f));
    float ang = (float)s * inv;
    float cs = __cosf(ang), sn = __sinf(ang);
    int base = s * HD + head * 128;
    float x1 = X[base + d], x2 = X[base + 64 + d];
    Y[base + d]      = f2bf(x1 * cs - x2 * sn);
    Y[base + 64 + d] = f2bf(x2 * cs + x1 * sn);
}

// ---------- C[M,N] = A[M,K] x B[N,K]^T, bf16 in, fp32 out ----------
// 256 threads = 8 waves; block tile 128x256, K-step 32; double-buffered LDS
// filled by async global->LDS copies. Each wave: 64x64 = 4x4 WMMA tiles.
#define BM 128
#define BN 256
#define BK 32
__global__ __launch_bounds__(256) void gemm_bf16_nt(const unsigned short* __restrict__ A,
                                                    const unsigned short* __restrict__ B,
                                                    float* __restrict__ C,
                                                    int M, int N, int K) {
    __shared__ alignas(16) unsigned short sA[2][BM * BK];
    __shared__ alignas(16) unsigned short sB[2][BN * BK];

    const int tid  = threadIdx.x;
    const int lane = tid & 31;
    const int wave = tid >> 5;          // 0..7
    const int wm   = wave >> 2;         // 0..1 : 64-row group
    const int wn   = wave & 3;          // 0..3 : 64-col group
    const int kh   = lane >> 4;         // K-half selector
    const int l16  = lane & 15;
    const int bm0  = blockIdx.y * BM;
    const int bn0  = blockIdx.x * BN;

    // A-tile loads: row tid>>1 (0..127), 16 halves at col (tid&1)*16
    const int arow = tid >> 1;
    const int acol = (tid & 1) << 4;
    // B-tile loads: row tid (0..255), full 32 halves
    const unsigned short* ga = A + (size_t)(bm0 + arow) * K + acol;
    const unsigned short* gb = B + (size_t)(bn0 + tid) * K;

    v8f acc[4][4] = {};

    auto issue_tile = [&](int buf, int k0) {
        async_b128(lds_off(&sA[buf][arow * BK + acol]),     ga + k0);
        async_b128(lds_off(&sA[buf][arow * BK + acol + 8]), ga + k0 + 8);
        async_b128(lds_off(&sB[buf][tid * BK + 0]),  gb + k0);
        async_b128(lds_off(&sB[buf][tid * BK + 8]),  gb + k0 + 8);
        async_b128(lds_off(&sB[buf][tid * BK + 16]), gb + k0 + 16);
        async_b128(lds_off(&sB[buf][tid * BK + 24]), gb + k0 + 24);
    };

    issue_tile(0, 0);
    int buf = 0;
    for (int k0 = 0; k0 < K; k0 += BK) {
        wait_async0();                    // our writes into buf landed
        __syncthreads();                  // everyone's writes into buf landed
        if (k0 + BK < K) {
            issue_tile(buf ^ 1, k0 + BK); // overlap next copy with this compute
            __builtin_prefetch(ga + k0 + 2 * BK, 0, 1);   // warm L2 two tiles out
            __builtin_prefetch(gb + k0 + 2 * BK, 0, 1);
        }

        // A fragments: lane l16 = row; elems 0..7 -> K=kh*8.., 8..15 -> K=16+kh*8..
        Frag fa[4];
#pragma unroll
        for (int mt = 0; mt < 4; ++mt) {
            const unsigned short* p = &sA[buf][(wm * 64 + mt * 16 + l16) * BK];
            fa[mt].u[0] = *(const uint4*)(p + kh * 8);
            fa[mt].u[1] = *(const uint4*)(p + 16 + kh * 8);
        }
#pragma unroll
        for (int nt = 0; nt < 4; ++nt) {
            // B fragment: lane l16 = output col; elems 0..7 -> K=kh*8.., 8..15 -> K=16+kh*8..
            Frag fb;
            const unsigned short* p = &sB[buf][(wn * 64 + nt * 16 + l16) * BK];
            fb.u[0] = *(const uint4*)(p + kh * 8);
            fb.u[1] = *(const uint4*)(p + 16 + kh * 8);
#pragma unroll
            for (int mt = 0; mt < 4; ++mt)
                acc[mt][nt] = __builtin_amdgcn_wmma_f32_16x16x32_bf16(
                    false, frag_cast(fa[mt]), false, frag_cast(fb),
                    (short)0, acc[mt][nt], false, false);
        }
        __syncthreads();
        buf ^= 1;
    }

    // C/D layout: VGPR r -> row r + 8*kh ; lane l16 -> col
#pragma unroll
    for (int mt = 0; mt < 4; ++mt)
#pragma unroll
        for (int nt = 0; nt < 4; ++nt) {
            float* cp = C + (size_t)(bm0 + wm * 64 + mt * 16 + kh * 8) * N
                          + (bn0 + wn * 64 + nt * 16 + l16);
#pragma unroll
            for (int r = 0; r < 8; ++r)
                cp[(size_t)r * N] = acc[mt][nt][r];
        }
}

// ---------- flash attention: one wave per (head, 16-query block) ----------
// Q [S,4096] bf16 (roped), K [S,1024] bf16 (roped), V [S,1024] bf16 -> O [S,4096] bf16
__global__ __launch_bounds__(32) void flash_attn(const unsigned short* __restrict__ Q,
                                                 const unsigned short* __restrict__ Kc,
                                                 const unsigned short* __restrict__ Vc,
                                                 unsigned short* __restrict__ O) {
    __shared__ alignas(16) unsigned short sP[16 * 32];     // probs, [row][key]
    __shared__ alignas(16) unsigned short sVT[128 * 32];   // V^T,  [d][key]

    const int lane = threadIdx.x & 31;
    const int kh   = lane >> 4;
    const int l16  = lane & 15;
    const int head = blockIdx.x >> 7;       // 0..31
    const int qb   = blockIdx.x & 127;      // 16-query block
    const int kvh  = head >> 2;             // KV head (GQA 4:1)
    const float scale = 0.08838834764831845f;  // 1/sqrt(128)

    // Q fragments: 4 chunks of K-dim 32; lane l16 = query row
    Frag fq[4];
    const unsigned short* qp = Q + (size_t)(qb * 16 + l16) * 4096 + head * 128;
#pragma unroll
    for (int c = 0; c < 4; ++c) {
        fq[c].u[0] = *(const uint4*)(qp + c * 32 + kh * 8);
        fq[c].u[1] = *(const uint4*)(qp + c * 32 + 16 + kh * 8);
    }

    v8f acc[8] = {};
    float mrow[8], lrow[8];
#pragma unroll
    for (int r = 0; r < 8; ++r) { mrow[r] = -3.0e38f; lrow[r] = 0.0f; }

    const int nkb = (qb >> 1) + 1;   // 32-key causal blocks
    for (int jb = 0; jb < nkb; ++jb) {
        const int k0 = jb * 32;

        // ---- scores: two 16x16 tiles over D=128 ----
        v8f sc[2] = {};
#pragma unroll
        for (int t = 0; t < 2; ++t)
#pragma unroll
            for (int c = 0; c < 4; ++c) {
                Frag fk;   // B-operand straight from global: d at kh*8.. and 16+kh*8..
                const unsigned short* kp = Kc + (size_t)(k0 + t * 16 + l16) * 1024
                                             + kvh * 128 + c * 32;
                fk.u[0] = *(const uint4*)(kp + kh * 8);
                fk.u[1] = *(const uint4*)(kp + 16 + kh * 8);
                sc[t] = __builtin_amdgcn_wmma_f32_16x16x32_bf16(
                    false, frag_cast(fq[c]), false, frag_cast(fk),
                    (short)0, sc[t], false, false);
            }

        // ---- stage V^T into LDS (lane = key row, coalesced global reads) ----
        const unsigned short* vp = Vc + (size_t)(k0 + lane) * 1024 + kvh * 128;
#pragma unroll
        for (int d = 0; d < 128; d += 8) {
            uint4 vv = *(const uint4*)(vp + d);
            const unsigned short* pe = (const unsigned short*)&vv;
#pragma unroll
            for (int e = 0; e < 8; ++e) sVT[(d + e) * 32 + lane] = pe[e];
        }

        // ---- streaming softmax update ----
        float p0v[8], p1v[8], alpha[8];
#pragma unroll
        for (int r = 0; r < 8; ++r) {
            int qrow = qb * 16 + kh * 8 + r;
            float s0 = sc[0][r] * scale;
            float s1 = sc[1][r] * scale;
            s0 = (k0 + l16      > qrow) ? -3.0e38f : s0;   // causal mask
            s1 = (k0 + 16 + l16 > qrow) ? -3.0e38f : s1;
            float mx = fmaxf(s0, s1);
#pragma unroll
            for (int o = 1; o < 16; o <<= 1) mx = fmaxf(mx, __shfl_xor(mx, o, 32));
            float mnew = fmaxf(mrow[r], mx);
            float a  = __expf(mrow[r] - mnew);
            float p0 = __expf(s0 - mnew);
            float p1 = __expf(s1 - mnew);
            float ps = p0 + p1;
#pragma unroll
            for (int o = 1; o < 16; o <<= 1) ps += __shfl_xor(ps, o, 32);
            lrow[r] = lrow[r] * a + ps;
            mrow[r] = mnew;
            alpha[r] = a; p0v[r] = p0; p1v[r] = p1;
        }
#pragma unroll
        for (int c = 0; c < 8; ++c)
#pragma unroll
            for (int r = 0; r < 8; ++r) acc[c][r] *= alpha[r];

        // ---- transpose P through LDS into A-fragment layout ----
#pragma unroll
        for (int r = 0; r < 8; ++r) {
            sP[(kh * 8 + r) * 32 + l16]      = f2bf(p0v[r]);
            sP[(kh * 8 + r) * 32 + 16 + l16] = f2bf(p1v[r]);
        }
        __syncthreads();
        Frag fp;
        {
            const unsigned short* pp = &sP[l16 * 32];
            fp.u[0] = *(const uint4*)(pp + kh * 8);
            fp.u[1] = *(const uint4*)(pp + 16 + kh * 8);
        }

        // ---- O += P (16x32) x V (32x128), 8 d-chunks ----
#pragma unroll
        for (int c = 0; c < 8; ++c) {
            Frag fv;   // lane l16 = d col; keys at kh*8.. and 16+kh*8..
            const unsigned short* pv = &sVT[(c * 16 + l16) * 32];
            fv.u[0] = *(const uint4*)(pv + kh * 8);
            fv.u[1] = *(const uint4*)(pv + 16 + kh * 8);
            acc[c] = __builtin_amdgcn_wmma_f32_16x16x32_bf16(
                false, frag_cast(fp), false, frag_cast(fv),
                (short)0, acc[c], false, false);
        }
        __syncthreads();
    }

    // ---- epilogue: O /= l, store bf16 ----
#pragma unroll
    for (int c = 0; c < 8; ++c)
#pragma unroll
        for (int r = 0; r < 8; ++r) {
            float o = acc[c][r] / lrow[r];
            int row = qb * 16 + kh * 8 + r;
            O[(size_t)row * 4096 + head * 128 + c * 16 + l16] = f2bf(o);
        }
}

// ---------- host launcher ----------
extern "C" void kernel_launch(void* const* d_in, const int* in_sizes, int n_in,
                              void* d_out, int out_size, void* d_ws, size_t ws_size,
                              hipStream_t stream) {
    (void)in_sizes; (void)n_in; (void)out_size; (void)ws_size;
    const float* X  = (const float*)d_in[0];
    const float* Wq = (const float*)d_in[1];
    const float* Wk = (const float*)d_in[2];
    const float* Wv = (const float*)d_in[3];
    const float* Wo = (const float*)d_in[4];
    float* out = (float*)d_out;

    const int S = 2048, H = 4096, KV = 1024;

    char* ws = (char*)d_ws;
    size_t off = 0;
    auto alloc = [&](size_t b) { char* p = ws + off; off += (b + 255) & ~(size_t)255; return p; };

    unsigned short* Xbf  = (unsigned short*)alloc((size_t)S * H * 2);
    unsigned short* Wqb  = (unsigned short*)alloc((size_t)H * H * 2);
    unsigned short* Wkb  = (unsigned short*)alloc((size_t)KV * H * 2);
    unsigned short* Wvb  = (unsigned short*)alloc((size_t)KV * H * 2);
    unsigned short* Wob  = (unsigned short*)alloc((size_t)H * H * 2);
    float*          Qf   = (float*)alloc((size_t)S * H * 4);
    float*          Kf   = (float*)alloc((size_t)S * KV * 4);
    float*          Vf   = (float*)alloc((size_t)S * KV * 4);
    unsigned short* Qbf  = (unsigned short*)alloc((size_t)S * H * 2);
    unsigned short* Kbf  = (unsigned short*)alloc((size_t)S * KV * 2);
    unsigned short* Vbf  = (unsigned short*)alloc((size_t)S * KV * 2);
    unsigned short* Abf  = (unsigned short*)alloc((size_t)S * H * 2);

    auto cvt = [&](const float* src, unsigned short* dst, size_t n) {
        cvt_bf16<<<dim3((unsigned)((n / 4 + 255) / 256)), 256, 0, stream>>>(src, dst, (int)n);
    };

    // bf16 conversions (one-time weight+activation quantize)
    cvt(X,  Xbf, (size_t)S * H);
    cvt(Wq, Wqb, (size_t)H * H);
    cvt(Wk, Wkb, (size_t)KV * H);
    cvt(Wv, Wvb, (size_t)KV * H);
    cvt(Wo, Wob, (size_t)H * H);

    // projections: y = x @ W^T
    gemm_bf16_nt<<<dim3(H / BN,  S / BM), 256, 0, stream>>>(Xbf, Wqb, Qf, S, H,  H);
    gemm_bf16_nt<<<dim3(KV / BN, S / BM), 256, 0, stream>>>(Xbf, Wkb, Kf, S, KV, H);
    gemm_bf16_nt<<<dim3(KV / BN, S / BM), 256, 0, stream>>>(Xbf, Wvb, Vf, S, KV, H);

    // RoPE(Q), RoPE(K), convert V
    rope_convert<<<dim3((S * (H / 2)  + 255) / 256), 256, 0, stream>>>(Qf, Qbf, S, H);
    rope_convert<<<dim3((S * (KV / 2) + 255) / 256), 256, 0, stream>>>(Kf, Kbf, S, KV);
    cvt(Vf, Vbf, (size_t)S * KV);

    // causal flash attention: 32 heads x 128 query-blocks, one wave each
    flash_attn<<<dim3(32 * 128), 32, 0, stream>>>(Qbf, Kbf, Vbf, Abf);

    // output projection -> fp32 result
    gemm_bf16_nt<<<dim3(H / BN, S / BM), 256, 0, stream>>>(Abf, Wob, out, S, H, H);
}
